// RSSM_71090298683495
// MI455X (gfx1250) — matrix operand order repaired and checked by reading, not compiled, not run on hardware
//
#include <hip/hip_runtime.h>
#include <hip/hip_bf16.h>
#include <math.h>

// ---------------------------------------------------------------------------
// RSSM scan for MI455X (gfx1250, wave32).
//   - bf16 WMMA (v_wmma_f32_16x16x32_bf16), f32 accumulation
//   - software-pipelined fragment loads (double buffered) to hide L2 latency
//   - async global->LDS staging (GLOBAL_LOAD_ASYNC_TO_LDS_B128) where the
//     stage is a pure byte copy, guarded by __has_builtin
//   - 4 batch groups x 16 WGs; per-step sync via device-scope sense barriers
//   - prior MLP = separate fully parallel WMMA kernel over B*T rows
// ---------------------------------------------------------------------------

typedef __attribute__((ext_vector_type(16))) __bf16 v16bf;
typedef __attribute__((ext_vector_type(8)))  float  v8f;
typedef __attribute__((ext_vector_type(4)))  int    v4i;

union FragU { uint4 q[2]; v16bf v; };

constexpr int BB = 64, TT = 256, HH = 1024, SS = 128, OO = 1024, AC = 32, XD = 160;
constexpr int LDA   = 1032;   // halves, 16B-aligned rows, padded
constexpr int LDACC = 264;    // floats
constexpr int WPG   = 16;     // workgroups per batch group

// output layout (floats): pm, ps, qm, qs [B,S,T]; h_seq [B,T,H]; z_seq [B,T,S]
constexpr size_t PM_OFF = 0;
constexpr size_t PS_OFF = 2097152;
constexpr size_t QM_OFF = 4194304;
constexpr size_t QS_OFF = 6291456;
constexpr size_t HS_OFF = 8388608;
constexpr size_t ZS_OFF = 25165824;

// workspace layout (bytes)
constexpr size_t WIHF_OFF  = 0;                                   // 192 t x 5 c
constexpr size_t WHHF_OFF  = WIHF_OFF  + 192ull *  5 * 512 * 2;   // 192 t x 32 c
constexpr size_t POW1F_OFF = WHHF_OFF  + 192ull * 32 * 512 * 2;   //  64 t x 64 c
constexpr size_t POW2F_OFF = POW1F_OFF +  64ull * 64 * 512 * 2;   //  16 t x 32 c
constexpr size_t PRW1F_OFF = POW2F_OFF +  16ull * 32 * 512 * 2;   //  32 t x 32 c
constexpr size_t PRW2F_OFF = PRW1F_OFF +  32ull * 32 * 512 * 2;   //  16 t x 16 c
constexpr size_t HBUF_OFF  = PRW2F_OFF +  16ull * 16 * 512 * 2;   // 2 x 64 x 1024 bf16
constexpr size_t XBUF_OFF  = HBUF_OFF  +   2ull * 64 * 1024 * 2;  // 2 x 64 x 160 bf16
constexpr size_t YBUF_OFF  = XBUF_OFF  +   2ull * 64 * 160 * 2;   // 64 x 1024 bf16
constexpr size_t H32_OFF   = YBUF_OFF  +  64ull * 1024 * 2;       // 64 x 1024 f32
constexpr size_t BAR_OFF   = H32_OFF   +  64ull * 1024 * 4;       // 4 x 16 uint

// ---- optional CDNA5 async global->LDS copy path -----------------------------
#if defined(__has_builtin)
#if __has_builtin(__builtin_amdgcn_global_load_async_to_lds_b128) && \
    __has_builtin(__builtin_amdgcn_s_wait_asynccnt)
#define HAVE_ASYNC_LDS 1
#endif
#endif
#ifndef HAVE_ASYNC_LDS
#define HAVE_ASYNC_LDS 0
#endif

#if HAVE_ASYNC_LDS
typedef __attribute__((address_space(1))) v4i* as1_v4i;   // "__device__" ptr
typedef __attribute__((address_space(3))) v4i* as3_v4i;   // "__shared__" ptr
// Copy one 16-byte chunk global->LDS without touching VGPR data paths.
__device__ __forceinline__ void async_b128(const void* g, void* l) {
  // AS1 pointer is numerically identical to the generic global pointer;
  // low 32 bits of a generic LDS pointer are the LDS offset.
  as1_v4i gp = (as1_v4i)(unsigned long long)(__UINTPTR_TYPE__)g;
  as3_v4i lp = (as3_v4i)(unsigned)(__UINTPTR_TYPE__)l;
  __builtin_amdgcn_global_load_async_to_lds_b128(gp, lp, 0, 0);
}
#endif

__device__ __forceinline__ float softplusf(float x) {
  return (x > 20.f) ? x : log1pf(expf(x));
}
__device__ __forceinline__ float sigmf(float x) { return 1.f / (1.f + expf(-x)); }

// Accumulate WMMA tiles over K-chunks [c0,c1), software-pipelined 2 deep so
// the loads for chunk c+1 are in flight while chunk c's WMMA executes.
// A: LDS, row-major bf16, stride lda halves. A-fragment k order used here (contiguous, matches B):
//   lanes 0-15 : row m=lane,    K = c*32 + {0..15}
//   lanes 16-31: row m=lane-16, K = c*32 + {16..31}
// B: global, pre-swizzled frag order: [chunk][lane][16 halves], lane n=l&15,
//   K = c*32 + (l<16 ? 0 : 16) + {0..15}.
__device__ __forceinline__ v8f wmma_accum_range(const __bf16* sA, int lda,
                                                const __bf16* Bfrag,
                                                int c0, int c1, int lane, v8f acc) {
  const int m  = lane & 15;
  const int kb = (lane < 16) ? 0 : 16;
  const __bf16* ap = sA + m * lda + c0 * 32 + kb;
  const __bf16* bp = Bfrag + (size_t)c0 * 512 + (size_t)lane * 16;
  FragU a0, b0;
  a0.q[0] = *(const uint4*)(ap);
  a0.q[1] = *(const uint4*)(ap + 8);
  b0.q[0] = *(const uint4*)(bp);
  b0.q[1] = *(const uint4*)(bp + 8);
#pragma unroll
  for (int c = c0; c < c1 - 1; ++c) {
    FragU a1, b1;
    ap += 32;
    bp += 512;
    a1.q[0] = *(const uint4*)(ap);
    a1.q[1] = *(const uint4*)(ap + 8);
    b1.q[0] = *(const uint4*)(bp);
    b1.q[1] = *(const uint4*)(bp + 8);
    acc = __builtin_amdgcn_wmma_f32_16x16x32_bf16(false, a0.v, false, b0.v,
                                                  (short)0, acc, false, false);
    a0 = a1;
    b0 = b1;
  }
  acc = __builtin_amdgcn_wmma_f32_16x16x32_bf16(false, a0.v, false, b0.v,
                                                (short)0, acc, false, false);
  return acc;
}

// C-fragment layout: VGPR i, lane l -> M = i + (l<16?0:8), N = l&15.
__device__ __forceinline__ void scatter_add(float* sAcc, int ldacc, int col0,
                                            int lane, v8f acc) {
  const int mB = (lane < 16) ? 0 : 8;
  const int n  = lane & 15;
#pragma unroll
  for (int i = 0; i < 8; ++i)
    atomicAdd(&sAcc[(mB + i) * ldacc + col0 + n], acc[i]);  // ds_add_f32
}

// Device-scope sense barrier over WPG workgroups. bar[0]=counter, bar[1]=gen.
__device__ void group_barrier(unsigned* bar, int tid) {
  __syncthreads();
  if (tid == 0) {
    __threadfence();
    unsigned gen  = __hip_atomic_load(&bar[1], __ATOMIC_RELAXED, __HIP_MEMORY_SCOPE_AGENT);
    unsigned prev = __hip_atomic_fetch_add(&bar[0], 1u, __ATOMIC_ACQ_REL, __HIP_MEMORY_SCOPE_AGENT);
    if (prev == WPG - 1) {
      __hip_atomic_store(&bar[0], 0u, __ATOMIC_RELAXED, __HIP_MEMORY_SCOPE_AGENT);
      __hip_atomic_fetch_add(&bar[1], 1u, __ATOMIC_RELEASE, __HIP_MEMORY_SCOPE_AGENT);
    } else {
      while (__hip_atomic_load(&bar[1], __ATOMIC_ACQUIRE, __HIP_MEMORY_SCOPE_AGENT) == gen)
        __builtin_amdgcn_s_sleep(2);
    }
    __threadfence();
  }
  __syncthreads();
}

// fp32 [N,K] row-major -> bf16 WMMA-B fragment order [tile][chunk][lane][16].
__global__ void convert_wfrag(const float* __restrict__ src, __bf16* __restrict__ dst,
                              int Ktot, int Kchunks, long total) {
  long i = (long)blockIdx.x * blockDim.x + threadIdx.x;
  if (i >= total) return;
  int  i16  = (int)(i & 15);
  int  lane = (int)((i >> 4) & 31);
  long rest = i >> 9;
  int  chunk = (int)(rest % Kchunks);
  long tile  = rest / Kchunks;
  int  n = (int)tile * 16 + (lane & 15);
  int  k = chunk * 32 + ((lane < 16) ? 0 : 16) + i16;
  dst[i] = (__bf16)src[(size_t)n * Ktot + k];
}

__global__ void init_bars(unsigned* bars) {
  int t = threadIdx.x;
  if (t < 8) bars[(t >> 1) * 16 + (t & 1)] = 0u;
}

// ---------------------------------------------------------------------------
// Sequential scan: GRU + posterior MLP.  grid = 64 WGs (4 groups x 16), 256 thr.
// ---------------------------------------------------------------------------
__global__ __launch_bounds__(256, 1) void rssm_scan(
    const float* __restrict__ obs, const float* __restrict__ action,
    const unsigned char* __restrict__ resets, const float* __restrict__ noise,
    const float* __restrict__ b_ih, const float* __restrict__ b_hh,
    const float* __restrict__ po_b1, const float* __restrict__ po_b2,
    const __bf16* __restrict__ wihf, const __bf16* __restrict__ whhf,
    const __bf16* __restrict__ pow1f, const __bf16* __restrict__ pow2f,
    __bf16* __restrict__ hbuf, __bf16* __restrict__ xbuf,
    __bf16* __restrict__ ybuf, float* __restrict__ h32,
    unsigned* __restrict__ bars, float* __restrict__ out) {
  __shared__ __attribute__((aligned(16))) __bf16 sA[16 * LDA];
  __shared__ float sAcc[16 * LDACC];

  const int tid  = threadIdx.x;
  const int lane = tid & 31;
  const int w    = tid >> 5;             // wave 0..7
  const int bg   = blockIdx.x / WPG;     // batch group 0..3
  const int wg   = blockIdx.x % WPG;     // N-slice within group
  unsigned* bar  = bars + bg * 16;

  // ---- init carries to zero (own slice) ----
  for (int e = tid; e < 16 * 64; e += 256) {
    int m = e >> 6, c = e & 63;
    int b = bg * 16 + m, hc = wg * 64 + c;
    hbuf[(size_t)b * HH + hc] = (__bf16)0.f;   // ping buffer 0
    h32[(size_t)b * HH + hc]  = 0.f;
  }
  if (wg == 0)
    for (int e = tid; e < 16 * XD; e += 256) {
      int m = e / XD, k = e % XD;
      xbuf[(size_t)(bg * 16 + m) * XD + k] = (__bf16)0.f;  // ping 0
    }
  group_barrier(bar, tid);

  for (int t = 0; t < TT; ++t) {
    const int cur = t & 1, nxt = cur ^ 1;

    // ===== phase 1: GRU gates =====
    for (int e = tid; e < 16 * 256; e += 256) sAcc[(e >> 8) * LDACC + (e & 255)] = 0.f;
    // stage x = [z(masked), a] (carry z stored raw; mask at stage)
    for (int e = tid; e < 16 * XD; e += 256) {
      int m = e / XD, k = e % XD, b = bg * 16 + m;
      float v = (float)xbuf[(size_t)cur * BB * XD + (size_t)b * XD + k];
      if (k < SS) v *= (resets[b] ? 0.f : 1.f);
      sA[m * LDA + k] = (__bf16)v;
    }
    __syncthreads();
    // gi: 12 (gate,tile) jobs, K=160 (5 chunks)
    for (int jj = 0; jj < 2; ++jj) {
      int j = w + jj * 8;
      if (j < 12) {
        int g = j >> 2, tt2 = j & 3;
        int tile = g * 64 + wg * 4 + tt2;
        v8f acc = {};
        acc = wmma_accum_range(sA, LDA, wihf + (size_t)tile * 5 * 512, 0, 5, lane, acc);
        scatter_add(sAcc, LDACC, ((g == 2) ? 128 : g * 64) + tt2 * 16, lane, acc);
      }
    }
    __syncthreads();
    // stage h (masked copy -> must go through VALU)
    for (int e = tid; e < 16 * HH; e += 256) {
      int m = e >> 10, k = e & 1023, b = bg * 16 + m;
      float keep = resets[b] ? 0.f : 1.f;
      sA[m * LDA + k] =
          (__bf16)((float)hbuf[(size_t)cur * BB * HH + (size_t)b * HH + k] * keep);
    }
    __syncthreads();
    // gh: 24 (gate,tile,khalf) jobs, 3 per wave, K=1024 (32 chunks)
    for (int ssj = 0; ssj < 3; ++ssj) {
      int s2 = w + ssj * 8;
      int j = s2 >> 1, kh = s2 & 1;
      int g = j >> 2, tt2 = j & 3;
      int tile = g * 64 + wg * 4 + tt2;
      v8f acc = {};
      acc = wmma_accum_range(sA, LDA, whhf + (size_t)tile * 32 * 512,
                             kh * 16, kh * 16 + 16, lane, acc);
      scatter_add(sAcc, LDACC, ((g == 2) ? 192 : g * 64) + tt2 * 16, lane, acc);
    }
    __syncthreads();
    // gate elementwise: acc = [r | u | gi_n | gh_n]
    for (int e = tid; e < 16 * 64; e += 256) {
      int m = e & 15, c = e >> 4;
      int b = bg * 16 + m, hc = wg * 64 + c;
      float r = sigmf(sAcc[m * LDACC + c] + b_ih[hc] + b_hh[hc]);
      float u = sigmf(sAcc[m * LDACC + 64 + c] + b_ih[HH + hc] + b_hh[HH + hc]);
      float n = tanhf(sAcc[m * LDACC + 128 + c] + b_ih[2 * HH + hc] +
                      r * (sAcc[m * LDACC + 192 + c] + b_hh[2 * HH + hc]));
      float keep = resets[b] ? 0.f : 1.f;
      float hold = h32[(size_t)b * HH + hc] * keep;
      float hn = (1.f - u) * n + u * hold;
      h32[(size_t)b * HH + hc] = hn;
      hbuf[(size_t)nxt * BB * HH + (size_t)b * HH + hc] = (__bf16)hn;
      out[HS_OFF + (size_t)b * TT * HH + (size_t)t * HH + hc] = hn;
    }
    group_barrier(bar, tid);  // h_new visible group-wide

    // ===== phase 2: posterior layer1, y1 = relu([h|obs] @ W1^T + b1) =====
    for (int e = tid; e < 16 * 64; e += 256) sAcc[(e >> 6) * LDACC + (e & 63)] = 0.f;
    __syncthreads();
    for (int pass = 0; pass < 2; ++pass) {
      if (pass == 0) {
        // pure byte copy hbuf[nxt] -> sA: async DMA path if available
#if HAVE_ASYNC_LDS
        for (int e = tid; e < 16 * (HH / 8); e += 256) {  // 16B chunks
          int m = e >> 7, c = e & 127;
          async_b128(hbuf + (size_t)nxt * BB * HH + (size_t)(bg * 16 + m) * HH + c * 8,
                     sA + m * LDA + c * 8);
        }
        __builtin_amdgcn_s_wait_asynccnt(0);
#else
        for (int e = tid; e < 16 * HH; e += 256) {
          int m = e >> 10, k = e & 1023, b = bg * 16 + m;
          sA[m * LDA + k] = hbuf[(size_t)nxt * BB * HH + (size_t)b * HH + k];
        }
#endif
      } else {
        for (int e = tid; e < 16 * OO; e += 256) {
          int m = e >> 10, k = e & 1023, b = bg * 16 + m;
          sA[m * LDA + k] =
              (__bf16)obs[(size_t)b * TT * OO + (size_t)t * OO + k];
        }
      }
      __syncthreads();
      int tt2 = w >> 1, kh = w & 1;
      int tile = wg * 4 + tt2;
      v8f acc = {};
      acc = wmma_accum_range(sA, LDA,
                             pow1f + ((size_t)tile * 64 + (size_t)pass * 32) * 512,
                             kh * 16, kh * 16 + 16, lane, acc);
      scatter_add(sAcc, LDACC, tt2 * 16, lane, acc);
      __syncthreads();
    }
    for (int e = tid; e < 16 * 64; e += 256) {
      int m = e & 15, c = e >> 4;
      int b = bg * 16 + m, nc = wg * 64 + c;
      float y = sAcc[m * LDACC + c] + po_b1[nc];
      ybuf[(size_t)b * HH + nc] = (__bf16)(y > 0.f ? y : 0.f);
    }
    group_barrier(bar, tid);  // y1 visible group-wide

    // ===== phase 3: posterior layer2 + rsample (WGs 0..7) =====
    if (wg < 8) {
      for (int e = tid; e < 16 * 32; e += 256) sAcc[(e >> 5) * LDACC + (e & 31)] = 0.f;
      // pure byte copy ybuf -> sA: async DMA path if available
#if HAVE_ASYNC_LDS
      for (int e = tid; e < 16 * (HH / 8); e += 256) {
        int m = e >> 7, c = e & 127;
        async_b128(ybuf + (size_t)(bg * 16 + m) * HH + c * 8, sA + m * LDA + c * 8);
      }
      __builtin_amdgcn_s_wait_asynccnt(0);
#else
      for (int e = tid; e < 16 * HH; e += 256) {
        int m = e >> 10, k = e & 1023, b = bg * 16 + m;
        sA[m * LDA + k] = ybuf[(size_t)b * HH + k];
      }
#endif
      __syncthreads();
      int t2 = w & 1, kq = w >> 1;                // tile {qm,qs}, K quarter
      int tile = (t2 == 0) ? wg : (8 + wg);
      v8f acc = {};
      acc = wmma_accum_range(sA, LDA, pow2f + (size_t)tile * 32 * 512,
                             kq * 8, kq * 8 + 8, lane, acc);
      scatter_add(sAcc, LDACC, t2 * 16, lane, acc);
      __syncthreads();
      {
        int m = tid >> 4, sc = tid & 15;          // 256 threads = 16x16
        int b = bg * 16 + m, s = wg * 16 + sc;
        float qm  = sAcc[m * LDACC + sc] + po_b2[s];
        float qs  = softplusf(sAcc[m * LDACC + 16 + sc] + po_b2[SS + s]) + 1e-6f;
        float nz  = noise[(size_t)b * TT * SS + (size_t)t * SS + s];
        float z   = qm + qs * nz;
        out[QM_OFF + (size_t)b * SS * TT + (size_t)s * TT + t] = qm;
        out[QS_OFF + (size_t)b * SS * TT + (size_t)s * TT + t] = qs;
        out[ZS_OFF + (size_t)b * TT * SS + (size_t)t * SS + s] = z;
        xbuf[(size_t)nxt * BB * XD + (size_t)b * XD + s] = (__bf16)z;
      }
    } else if (wg == 8 && t + 1 < TT) {
      for (int e = tid; e < 16 * AC; e += 256) {
        int m = e >> 5, c = e & 31, b = bg * 16 + m;
        xbuf[(size_t)nxt * BB * XD + (size_t)b * XD + SS + c] =
            (__bf16)action[(size_t)b * TT * AC + (size_t)t * AC + c];
      }
    }
    group_barrier(bar, tid);  // z/x ready for next step
  }
}

// ---------------------------------------------------------------------------
// Prior MLP: fully parallel over B*T rows. grid = 1024 WGs x 256 thr.
// ---------------------------------------------------------------------------
__global__ __launch_bounds__(256, 1) void rssm_prior(
    const float* __restrict__ hseq, const float* __restrict__ pr_b1,
    const float* __restrict__ pr_b2, const __bf16* __restrict__ prw1f,
    const __bf16* __restrict__ prw2f, float* __restrict__ out) {
  __shared__ __attribute__((aligned(16))) __bf16 sA[16 * LDA];
  __shared__ __attribute__((aligned(16))) __bf16 sY[16 * 520];
  const int tid = threadIdx.x, lane = tid & 31, w = tid >> 5;
  const int row0 = blockIdx.x * 16;

  for (int e = tid; e < 16 * HH; e += 256) {
    int m = e >> 10, k = e & 1023;
    sA[m * LDA + k] = (__bf16)hseq[(size_t)(row0 + m) * HH + k];
  }
  __syncthreads();
  // layer1: 32 n-tiles (N=512), K=1024 (32 chunks); relu into sY
  for (int j = 0; j < 4; ++j) {
    int tile = w + j * 8;
    v8f acc = {};
    acc = wmma_accum_range(sA, LDA, prw1f + (size_t)tile * 32 * 512, 0, 32, lane, acc);
    int n = tile * 16 + (lane & 15);
    int mB = (lane < 16) ? 0 : 8;
    float bias = pr_b1[n];
#pragma unroll
    for (int i = 0; i < 8; ++i) {
      float y = acc[i] + bias;
      sY[(mB + i) * 520 + n] = (__bf16)(y > 0.f ? y : 0.f);
    }
  }
  __syncthreads();
  // layer2: 16 n-tiles (N=256), K=512 (16 chunks); pm / softplus(ps)
  for (int j = 0; j < 2; ++j) {
    int tile = w + j * 8;
    v8f acc = {};
    acc = wmma_accum_range(sY, 520, prw2f + (size_t)tile * 16 * 512, 0, 16, lane, acc);
    int n = tile * 16 + (lane & 15);
    int mB = (lane < 16) ? 0 : 8;
    float bias = pr_b2[n];
#pragma unroll
    for (int i = 0; i < 8; ++i) {
      int r = row0 + mB + i, b = r / TT, t = r % TT;
      float v = acc[i] + bias;
      if (n < SS)
        out[PM_OFF + (size_t)b * SS * TT + (size_t)n * TT + t] = v;
      else
        out[PS_OFF + (size_t)b * SS * TT + (size_t)(n - SS) * TT + t] =
            softplusf(v) + 1e-6f;
    }
  }
}

extern "C" void kernel_launch(void* const* d_in, const int* in_sizes, int n_in,
                              void* d_out, int out_size, void* d_ws, size_t ws_size,
                              hipStream_t stream) {
  (void)in_sizes; (void)n_in; (void)out_size; (void)ws_size;
  const float* obs    = (const float*)d_in[0];
  const float* action = (const float*)d_in[1];
  const unsigned char* resets = (const unsigned char*)d_in[2];  // jnp.bool_ = 1 byte
  const float* noise  = (const float*)d_in[3];
  const float* w_ih   = (const float*)d_in[4];
  const float* w_hh   = (const float*)d_in[5];
  const float* b_ih   = (const float*)d_in[6];
  const float* b_hh   = (const float*)d_in[7];
  const float* pr_w1  = (const float*)d_in[8];
  const float* pr_b1  = (const float*)d_in[9];
  const float* pr_w2  = (const float*)d_in[10];
  const float* pr_b2  = (const float*)d_in[11];
  const float* po_w1  = (const float*)d_in[12];
  const float* po_b1  = (const float*)d_in[13];
  const float* po_w2  = (const float*)d_in[14];
  const float* po_b2  = (const float*)d_in[15];

  char* ws = (char*)d_ws;
  __bf16* wihf  = (__bf16*)(ws + WIHF_OFF);
  __bf16* whhf  = (__bf16*)(ws + WHHF_OFF);
  __bf16* pow1f = (__bf16*)(ws + POW1F_OFF);
  __bf16* pow2f = (__bf16*)(ws + POW2F_OFF);
  __bf16* prw1f = (__bf16*)(ws + PRW1F_OFF);
  __bf16* prw2f = (__bf16*)(ws + PRW2F_OFF);
  __bf16* hbuf  = (__bf16*)(ws + HBUF_OFF);
  __bf16* xbuf  = (__bf16*)(ws + XBUF_OFF);
  __bf16* ybuf  = (__bf16*)(ws + YBUF_OFF);
  float*  h32   = (float*)(ws + H32_OFF);
  unsigned* bars = (unsigned*)(ws + BAR_OFF);
  float* out = (float*)d_out;

  init_bars<<<1, 32, 0, stream>>>(bars);

  auto conv = [&](const float* src, __bf16* dst, int K, int Kc, long total) {
    convert_wfrag<<<(int)((total + 255) / 256), 256, 0, stream>>>(src, dst, K, Kc, total);
  };
  conv(w_ih,  wihf,  XD,   5, 192L *  5 * 512);
  conv(w_hh,  whhf,  HH,  32, 192L * 32 * 512);
  conv(po_w1, pow1f, 2048, 64, 64L * 64 * 512);
  conv(po_w2, pow2f, HH,  32, 16L * 32 * 512);
  conv(pr_w1, prw1f, HH,  32, 32L * 32 * 512);
  conv(pr_w2, prw2f, 512, 16, 16L * 16 * 512);

  rssm_scan<<<64, 256, 0, stream>>>(obs, action, resets, noise, b_ih, b_hh,
                                    po_b1, po_b2, wihf, whhf, pow1f, pow2f,
                                    hbuf, xbuf, ybuf, h32, bars, out);

  rssm_prior<<<BB * TT / 16, 256, 0, stream>>>(out + HS_OFF, pr_b1, pr_b2,
                                               prw1f, prw2f, out);
}
